// AttentionMoE_498216206668
// MI455X (gfx1250) — compile-verified, matrix operand order checked by reading, not run on hardware
//
#include <hip/hip_runtime.h>
#include <hip/hip_bf16.h>
#include <math.h>

#define BATCH 512
#define DDIM  2048
#define NEXP  8
#define MT 32     // M tile (tokens)
#define NT 128    // N tile
#define KT 32     // K step (bf16 wmma depth)
#define KP 40     // padded LDS row length in bf16 elems (80B, 16B-aligned, conflict-free)

typedef __bf16 bf16_t;
typedef __attribute__((ext_vector_type(16))) __bf16 v16bf;
typedef __attribute__((ext_vector_type(8)))  __bf16 v8bf;
typedef __attribute__((ext_vector_type(8)))  float  v8f;

__device__ __forceinline__ v16bf pack16(v8bf lo, v8bf hi) {
    return __builtin_shufflevector(lo, hi, 0,1,2,3,4,5,6,7,8,9,10,11,12,13,14,15);
}

// ---------------------------------------------------------------- zero counters
__global__ void zero_kernel(int* __restrict__ counts, float* __restrict__ loadv,
                            float* __restrict__ impv) {
    int t = threadIdx.x;
    if (t < NEXP) { counts[t] = 0; loadv[t] = 0.0f; impv[t] = 0.0f; }
}

// ---------------------------------------------------------------- gating + routing
__global__ __launch_bounds__(256) void gate_kernel(
    const float* __restrict__ x, const float* __restrict__ noise,
    const float* __restrict__ wg, const float* __restrict__ wn,
    int* __restrict__ counts, float* __restrict__ loadv, float* __restrict__ impv,
    int* __restrict__ lists, int* __restrict__ slots, float* __restrict__ sgates)
{
    const int b = blockIdx.x, tid = threadIdx.x;
    float pcl[NEXP], pns[NEXP];
#pragma unroll
    for (int e = 0; e < NEXP; ++e) { pcl[e] = 0.f; pns[e] = 0.f; }
    for (int d = tid; d < DDIM; d += 256) {
        float xv = x[(size_t)b * DDIM + d];
#pragma unroll
        for (int e = 0; e < NEXP; ++e) {
            pcl[e] += xv * wg[d * NEXP + e];
            pns[e] += xv * wn[d * NEXP + e];
        }
    }
    __shared__ float sh[256];
    __shared__ float cl[NEXP], nsr[NEXP];
    for (int e = 0; e < NEXP; ++e) {
        sh[tid] = pcl[e]; __syncthreads();
        for (int s = 128; s > 0; s >>= 1) { if (tid < s) sh[tid] += sh[tid + s]; __syncthreads(); }
        if (tid == 0) cl[e] = sh[0];
        __syncthreads();
        sh[tid] = pns[e]; __syncthreads();
        for (int s = 128; s > 0; s >>= 1) { if (tid < s) sh[tid] += sh[tid + s]; __syncthreads(); }
        if (tid == 0) nsr[e] = sh[0];
        __syncthreads();
    }
    if (tid == 0) {
        float logit[NEXP], nstd[NEXP];
#pragma unroll
        for (int e = 0; e < NEXP; ++e) {
            float v = nsr[e];
            float sp = (v > 15.0f) ? v : log1pf(expf(v));   // softplus
            nstd[e] = sp + 0.01f;
            logit[e] = cl[e] + noise[b * NEXP + e] * nstd[e];
        }
        // top-3 selection (stable: first index wins ties, like lax.top_k)
        int   idx[3]; float val[3];
        bool used[NEXP] = {false,false,false,false,false,false,false,false};
        for (int t = 0; t < 3; ++t) {
            float best = -3.4e38f; int bi = 0;
            for (int e = 0; e < NEXP; ++e)
                if (!used[e] && logit[e] > best) { best = logit[e]; bi = e; }
            used[bi] = true; idx[t] = bi; val[t] = best;
        }
        // softmax over top-2
        float m  = val[0];
        float e0 = expf(val[0] - m), e1 = expf(val[1] - m);
        float inv = 1.0f / (e0 + e1);
        float g0 = e0 * inv, g1 = e1 * inv;
        // load loss: prob via ndtr
        float thr_in = val[2], thr_out = val[1];
#pragma unroll
        for (int e = 0; e < NEXP; ++e) {
            float thr = (logit[e] > thr_in) ? thr_in : thr_out;
            atomicAdd(&loadv[e], normcdff((cl[e] - thr) / nstd[e]));
        }
        atomicAdd(&impv[idx[0]], g0);
        atomicAdd(&impv[idx[1]], g1);
        // routing
        int p0 = atomicAdd(&counts[idx[0]], 1);
        lists[idx[0] * BATCH + p0] = b;
        slots[2 * b + 0] = idx[0] * BATCH + p0;  sgates[2 * b + 0] = g0;
        int p1 = atomicAdd(&counts[idx[1]], 1);
        lists[idx[1] * BATCH + p1] = b;
        slots[2 * b + 1] = idx[1] * BATCH + p1;  sgates[2 * b + 1] = g1;
    }
}

// ---------------------------------------------------------------- aux loss finalize
__global__ void loss_kernel(const float* __restrict__ loadv, const float* __restrict__ impv,
                            float* __restrict__ out)
{
    if (threadIdx.x == 0) {
        float acc = 0.0f;
        for (int which = 0; which < 2; ++which) {
            const float* v = which ? loadv : impv;
            float m = 0.f;
            for (int e = 0; e < NEXP; ++e) m += v[e];
            m *= (1.0f / NEXP);
            float s = 0.f;
            for (int e = 0; e < NEXP; ++e) { float d = v[e] - m; s += d * d; }
            s *= (1.0f / (NEXP - 1));                     // ddof=1
            acc += s / (m * m + 1e-10f);
        }
        out[(size_t)BATCH * DDIM] = acc * 0.01f;
    }
}

// ---------------------------------------------------------------- bf16 WMMA expert GEMM
// Software double-buffered through registers: global loads for slab k+1 are issued
// before the barrier of slab k, so s_wait_loadcnt lands after the WMMA work.
// mode 0:  out_h  = tanh( X_e @ W1[e] + b1[e] )   (A = fp32 gathered tokens, out bf16)
// mode 1:  out_s  =        H_e @ W2[e] + b2[e]    (A = bf16 h rows,        out fp32)
__global__ __launch_bounds__(256) void expert_gemm(
    const float* __restrict__ W, const float* __restrict__ bias,
    const float* __restrict__ Af32, const bf16_t* __restrict__ Abf,
    const int* __restrict__ counts, const int* __restrict__ lists,
    bf16_t* __restrict__ outh, float* __restrict__ outs, int mode)
{
    const int e   = blockIdx.z;
    const int cnt = counts[e];
    const int m0  = blockIdx.y * MT;
    if (m0 >= cnt) return;                       // predicated on live tokens
    const int nb  = blockIdx.x * NT;
    const int tid = threadIdx.x;

    __shared__ __align__(16) bf16_t Al[MT][KP];  // fragment-ready [m][k]
    __shared__ __align__(16) bf16_t Bl[NT][KP];  // fragment-ready [n][k]

    const int lane = tid & 31;
    const int w    = tid >> 5;               // 8 waves
    const int wm   = (w >> 2) << 4;          // 0 / 16
    const int wn   = (w & 3) << 5;           // 0 / 32 / 64 / 96
    const int lm   = lane & 15;
    const int kb   = (lane >> 4) << 3;       // 0 / 8 (ISA 16-bit A/B lane K layout)

    v8f acc0 = {0.f,0.f,0.f,0.f,0.f,0.f,0.f,0.f};
    v8f acc1 = {0.f,0.f,0.f,0.f,0.f,0.f,0.f,0.f};

    // A staging assignment: one 4-elem chunk per thread
    const int am = tid >> 3;                 // row 0..31
    const int ac = (tid & 7) << 2;           // col 0..28
    int pA = m0 + am; if (pA > cnt - 1) pA = cnt - 1;   // clamp partial tile
    const float*  arow32 = (mode == 0) ? (Af32 + (size_t)lists[e * BATCH + pA] * DDIM) : nullptr;
    const bf16_t* arow16 = (mode == 1) ? (Abf + ((size_t)e * BATCH + pA) * DDIM) : nullptr;

    // B staging assignment: 4 float4 per thread
    const size_t wbase = (size_t)e * DDIM * DDIM + nb;
    int bk[4], bc[4];
#pragma unroll
    for (int i = 0; i < 4; ++i) {
        int slot = tid + (i << 8);
        bk[i] = slot >> 5;                   // k row 0..31
        bc[i] = (slot & 31) << 2;            // n col 0..124
    }

    float4 aReg;  uint2 aRegH;  float4 bReg[4];

    auto issue_loads = [&](int kk) {
        if (mode == 0) aReg  = *(const float4*)(arow32 + kk + ac);
        else           aRegH = *(const uint2*)(arow16 + kk + ac);
#pragma unroll
        for (int i = 0; i < 4; ++i)
            bReg[i] = *(const float4*)(W + wbase + (size_t)(kk + bk[i]) * DDIM + bc[i]);
        if (kk + 2 * KT < DDIM) {            // prefetch slab k+2 into L2
#pragma unroll
            for (int i = 0; i < 4; ++i)
                __builtin_prefetch(W + wbase + (size_t)(kk + KT + bk[i]) * DDIM + bc[i], 0, 1);
        }
    };
    auto store_tiles = [&]() {
        if (mode == 0) {
            Al[am][ac + 0] = (bf16_t)aReg.x;  Al[am][ac + 1] = (bf16_t)aReg.y;
            Al[am][ac + 2] = (bf16_t)aReg.z;  Al[am][ac + 3] = (bf16_t)aReg.w;
        } else {
            *(uint2*)&Al[am][ac] = aRegH;    // 4 bf16 raw copy
        }
#pragma unroll
        for (int i = 0; i < 4; ++i) {
            Bl[bc[i] + 0][bk[i]] = (bf16_t)bReg[i].x;
            Bl[bc[i] + 1][bk[i]] = (bf16_t)bReg[i].y;
            Bl[bc[i] + 2][bk[i]] = (bf16_t)bReg[i].z;
            Bl[bc[i] + 3][bk[i]] = (bf16_t)bReg[i].w;
        }
    };

    issue_loads(0);
    for (int kk = 0; kk < DDIM; kk += KT) {
        store_tiles();                                   // consumes regs of slab kk
        if (kk + KT < DDIM) issue_loads(kk + KT);        // in flight across the WMMAs
        __syncthreads();

        // fragments per ISA 16-bit layouts: elems 0-7 = K kb..kb+7, 8-15 = K kb+16..kb+23
        v16bf a  = pack16(*(const v8bf*)&Al[wm + lm][kb],      *(const v8bf*)&Al[wm + lm][kb + 16]);
        v16bf b0 = pack16(*(const v8bf*)&Bl[wn + lm][kb],      *(const v8bf*)&Bl[wn + lm][kb + 16]);
        v16bf b1 = pack16(*(const v8bf*)&Bl[wn + 16 + lm][kb], *(const v8bf*)&Bl[wn + 16 + lm][kb + 16]);

        acc0 = __builtin_amdgcn_wmma_f32_16x16x32_bf16(false, a, false, b0, (short)0, acc0, false, false);
        acc1 = __builtin_amdgcn_wmma_f32_16x16x32_bf16(false, a, false, b1, (short)0, acc1, false, false);
        __syncthreads();
    }

    // ---- epilogue (C layout: VGPR r -> M = r (+8 for lanes 16-31), N = lane&15)
#pragma unroll
    for (int r = 0; r < 8; ++r) {
        int m = wm + r + ((lane >= 16) ? 8 : 0);
        int p = m0 + m;
        if (p < cnt) {
            int n0 = nb + wn + lm;
            int n1 = n0 + 16;
            size_t row = ((size_t)e * BATCH + p) * DDIM;
            float v0 = acc0[r] + bias[e * DDIM + n0];
            float v1 = acc1[r] + bias[e * DDIM + n1];
            if (mode == 0) {
                outh[row + n0] = (bf16_t)tanhf(v0);
                outh[row + n1] = (bf16_t)tanhf(v1);
            } else {
                outs[row + n0] = v0;
                outs[row + n1] = v1;
            }
        }
    }
}

// ---------------------------------------------------------------- softmax + combine + log
// Score rows staged once in LDS (16 KB/block) -> 1x global read instead of 3x.
__global__ __launch_bounds__(256) void combine_kernel(
    const float* __restrict__ scores, const float* __restrict__ x,
    const int* __restrict__ slots, const float* __restrict__ sgates,
    float* __restrict__ y)
{
    const int b = blockIdx.x, tid = threadIdx.x;
    const float* r0 = scores + (size_t)slots[2 * b + 0] * DDIM;
    const float* r1 = scores + (size_t)slots[2 * b + 1] * DDIM;
    const float g0 = sgates[2 * b + 0], g1 = sgates[2 * b + 1];

    __shared__ float row0[DDIM];   // 8 KB
    __shared__ float row1[DDIM];   // 8 KB
    __shared__ float sh[256];

    for (int d = tid; d < DDIM; d += 256) { row0[d] = r0[d]; row1[d] = r1[d]; }
    __syncthreads();

    float m0 = -3.4e38f, m1 = -3.4e38f;
    for (int d = tid; d < DDIM; d += 256) { m0 = fmaxf(m0, row0[d]); m1 = fmaxf(m1, row1[d]); }
    sh[tid] = m0; __syncthreads();
    for (int s = 128; s > 0; s >>= 1) { if (tid < s) sh[tid] = fmaxf(sh[tid], sh[tid + s]); __syncthreads(); }
    m0 = sh[0]; __syncthreads();
    sh[tid] = m1; __syncthreads();
    for (int s = 128; s > 0; s >>= 1) { if (tid < s) sh[tid] = fmaxf(sh[tid], sh[tid + s]); __syncthreads(); }
    m1 = sh[0]; __syncthreads();

    float s0 = 0.f, s1 = 0.f;
    for (int d = tid; d < DDIM; d += 256) { s0 += expf(row0[d] - m0); s1 += expf(row1[d] - m1); }
    sh[tid] = s0; __syncthreads();
    for (int s = 128; s > 0; s >>= 1) { if (tid < s) sh[tid] += sh[tid + s]; __syncthreads(); }
    s0 = sh[0]; __syncthreads();
    sh[tid] = s1; __syncthreads();
    for (int s = 128; s > 0; s >>= 1) { if (tid < s) sh[tid] += sh[tid + s]; __syncthreads(); }
    s1 = sh[0]; __syncthreads();

    const float i0 = 1.0f / s0, i1 = 1.0f / s1;
    for (int d = tid; d < DDIM; d += 256) {
        float xv = x[(size_t)b * DDIM + d];
        float a0 = expf(row0[d] - m0) * i0;
        float a1 = expf(row1[d] - m1) * i1;
        float c  = g0 * expf(a0 * xv) + g1 * expf(a1 * xv);
        if (c == 0.0f) c = 2.220446049250313e-16f;
        y[(size_t)b * DDIM + d] = logf(c);
    }
}

// ---------------------------------------------------------------- launcher
extern "C" void kernel_launch(void* const* d_in, const int* in_sizes, int n_in,
                              void* d_out, int out_size, void* d_ws, size_t ws_size,
                              hipStream_t stream)
{
    const float* x     = (const float*)d_in[0];
    const float* noise = (const float*)d_in[1];
    const float* wg    = (const float*)d_in[2];
    const float* wn    = (const float*)d_in[3];
    const float* W1    = (const float*)d_in[4];
    const float* b1    = (const float*)d_in[5];
    const float* W2    = (const float*)d_in[6];
    const float* b2    = (const float*)d_in[7];
    float* y = (float*)d_out;

    // workspace layout (~50.4 MB):
    char* ws = (char*)d_ws;
    int*    counts = (int*)(ws + 0);                    //  32 B
    float*  loadv  = (float*)(ws + 32);                 //  32 B
    float*  impv   = (float*)(ws + 64);                 //  32 B
    int*    lists  = (int*)(ws + 128);                  //  8*512*4 = 16 KB
    int*    slots  = (int*)(ws + 128 + 16384);          //  4 KB
    float*  sgates = (float*)(ws + 128 + 16384 + 4096); //  4 KB
    bf16_t* hbuf   = (bf16_t*)(ws + 32768);                               // 16 MB (bf16 h)
    float*  sbuf   = (float*)(ws + 32768 + (size_t)NEXP*BATCH*DDIM*2);    // 32 MB (fp32 scores)

    zero_kernel<<<1, 32, 0, stream>>>(counts, loadv, impv);
    gate_kernel<<<BATCH, 256, 0, stream>>>(x, noise, wg, wn, counts, loadv, impv,
                                           lists, slots, sgates);
    loss_kernel<<<1, 32, 0, stream>>>(loadv, impv, y);

    dim3 grid(DDIM / NT, BATCH / MT, NEXP);   // (16, 16, 8), predicated on counts
    expert_gemm<<<grid, 256, 0, stream>>>(W1, b1, x, nullptr, counts, lists, hbuf, nullptr, 0);
    expert_gemm<<<grid, 256, 0, stream>>>(W2, b2, nullptr, hbuf, counts, lists, hbuf, sbuf, 1);

    combine_kernel<<<BATCH, 256, 0, stream>>>(sbuf, x, slots, sgates, y);
}
